// ExtractorHead_18451179503856
// MI455X (gfx1250) — matrix operand order repaired and checked by reading, not compiled, not run on hardware
//
#include <hip/hip_runtime.h>
#include <hip/hip_bf16.h>

typedef __attribute__((ext_vector_type(16))) _Float16 v16h;
typedef __attribute__((ext_vector_type(8)))  _Float16 v8h;
typedef __attribute__((ext_vector_type(4)))  _Float16 v4h;
typedef __attribute__((ext_vector_type(8)))  float    v8f;

#define HDIM   128
#define RFEAT  50
#define KPADE  320          // 2*H + R = 306 padded to 10 K-chunks of 32
#define KTE    10           // K tiles for edge GEMM
#define KTN    4            // K tiles for node GEMM (K = 128)
#define NTILES 8            // 128 / 16 output column tiles
#define TEDGE  64           // edges per block (4 row tiles of 16)
#define CUTOFF 6.0f

// x * v_rcp_f32(1+exp(-x)) : one v_exp + one v_rcp instead of an IEEE div macro
__device__ __forceinline__ float silu_f(float x) {
    return x * __builtin_amdgcn_rcpf(1.0f + __expf(-x));
}

// ---------------------------------------------------------------------------
// Prep kernels
// ---------------------------------------------------------------------------

__global__ void zero_f32_kernel(float* __restrict__ p, int n) {
    for (int i = blockIdx.x * blockDim.x + threadIdx.x; i < n; i += gridDim.x * blockDim.x)
        p[i] = 0.0f;
}

// Pack a [Ksrc x 128] f32 weight matrix into WMMA-B per-lane layout, f16,
// K padded to Kpad (multiple of 32). tile = kt*8 + nt ; element j of lane maps
// to k = kt*32 + (j&7) + (lane>>4)*8 + (j>>3)*16, n = nt*16 + (lane&15).
__global__ void pack_w_kernel(const float* __restrict__ W, _Float16* __restrict__ out,
                              int Ksrc, int Kpad) {
    int idx = blockIdx.x * blockDim.x + threadIdx.x;
    int total = Kpad * HDIM;
    if (idx >= total) return;
    int j    = idx & 15;
    int lane = (idx >> 4) & 31;
    int tile = idx >> 9;
    int nt = tile & 7;
    int kt = tile >> 3;
    int half = lane >> 4;
    int n = nt * 16 + (lane & 15);
    int k = kt * 32 + (j & 7) + half * 8 + (j >> 3) * 16;
    float val = (k < Ksrc) ? W[(long)k * HDIM + n] : 0.0f;
    out[idx] = (_Float16)val;
}

__global__ void demb_f16_kernel(const float* __restrict__ t, _Float16* __restrict__ o, int n) {
    int i = blockIdx.x * blockDim.x + threadIdx.x;
    if (i < n) o[i] = (_Float16)t[i];
}

__global__ void embed_kernel(const int* __restrict__ z, const float* __restrict__ emb,
                             float* __restrict__ v, _Float16* __restrict__ vh, int N) {
    int idx = blockIdx.x * blockDim.x + threadIdx.x;
    if (idx >= N * HDIM) return;
    int n = idx >> 7, f = idx & 127;
    float val = emb[(long)z[n] * HDIM + f];
    v[idx]  = val;
    vh[idx] = (_Float16)val;
}

__global__ void pos_copy_kernel(const float* __restrict__ pos, float* __restrict__ out, int n3) {
    int i = blockIdx.x * blockDim.x + threadIdx.x;
    if (i < n3) out[i] = pos[i];
}

__global__ void geom_kernel(const float* __restrict__ pos, const float* __restrict__ cell,
                            const int* __restrict__ eidx, const int* __restrict__ coff,
                            const int* __restrict__ ebatch,
                            float* __restrict__ dist_out, float* __restrict__ env,
                            int* __restrict__ didx, int E) {
    int e = blockIdx.x * blockDim.x + threadIdx.x;
    if (e >= E) return;
    int r = eidx[e], c = eidx[E + e];
    const float* C = cell + (long)ebatch[e] * 9;
    float o0 = (float)coff[e * 3 + 0];
    float o1 = (float)coff[e * 3 + 1];
    float o2 = (float)coff[e * 3 + 2];
    float ox = o0 * C[0] + o1 * C[3] + o2 * C[6];
    float oy = o0 * C[1] + o1 * C[4] + o2 * C[7];
    float oz = o0 * C[2] + o1 * C[5] + o2 * C[8];
    float dx = pos[r * 3 + 0] - pos[c * 3 + 0] + ox;
    float dy = pos[r * 3 + 1] - pos[c * 3 + 1] + oy;
    float dz = pos[r * 3 + 2] - pos[c * 3 + 2] + oz;
    float d = sqrtf(dx * dx + dy * dy + dz * dz);
    dist_out[e] = d;
    int di = (int)d;
    di = di < 0 ? 0 : (di > RFEAT - 1 ? RFEAT - 1 : di);
    didx[e] = di;
    float dc = fminf(fmaxf(d, 0.0f), CUTOFF);
    env[e] = 0.5f * (__cosf(3.14159265358979323846f * dc / CUTOFF) + 1.0f);
}

// ---------------------------------------------------------------------------
// Edge layer: e = silu([v[row]|v[col]|demb] @ We + be) * env ; agg += scatter
// 256 threads = 8 waves, 64 edges per block (4 row tiles x 8 col tiles;
// wave w owns column tile w and all 4 row tiles -> each B register load
// feeds 4 WMMAs).
// ---------------------------------------------------------------------------
__global__ void __launch_bounds__(256)
edge_layer_kernel(const _Float16* __restrict__ vh, const _Float16* __restrict__ dembh,
                  const int* __restrict__ didx, const float* __restrict__ env,
                  const int* __restrict__ erow, const int* __restrict__ ecol,
                  const _Float16* __restrict__ Wp,   // packed [KTE*8*512]
                  const float* __restrict__ bias,    // [128]
                  float* __restrict__ agg, int E) {
    __shared__ _Float16 sm[TEDGE * KPADE];            // 40 KB message staging
    __shared__ int   s_col[TEDGE];
    __shared__ float s_env[TEDGE];
    const int tid = threadIdx.x;
    const int e0  = blockIdx.x * TEDGE;

    // stage edge metadata once
    __shared__ int s_row[TEDGE], s_didx[TEDGE];
    if (tid < TEDGE) {
        int e = e0 + tid;
        bool ok = e < E;
        s_row[tid]  = ok ? erow[e] : 0;
        s_col[tid]  = ok ? ecol[e] : 0;
        s_didx[tid] = ok ? didx[e] : 0;
        s_env[tid]  = ok ? env[e]  : 0.0f;
    }
    __syncthreads();

    // vectorized fill: TEDGE edges x 40 vec8 slots, 16-byte transfers
    for (int idx = tid; idx < TEDGE * (KPADE / 8); idx += 256) {
        int el   = idx / (KPADE / 8);
        int f0   = (idx - el * (KPADE / 8)) * 8;
        int e    = e0 + el;
        uint4 val = make_uint4(0u, 0u, 0u, 0u);
        if (e < E) {
            if (f0 < HDIM) {
                val = *(const uint4*)(vh + (long)s_row[el] * HDIM + f0);
            } else if (f0 < 2 * HDIM) {
                val = *(const uint4*)(vh + (long)s_col[el] * HDIM + (f0 - HDIM));
            } else {
                _Float16 tmp[8];
                const _Float16* dsrc = dembh + s_didx[el] * RFEAT;
#pragma unroll
                for (int i = 0; i < 8; ++i) {
                    int fe = f0 - 2 * HDIM + i;
                    tmp[i] = (fe < RFEAT) ? dsrc[fe] : (_Float16)0.0f;
                }
                val = *(const uint4*)tmp;
            }
        }
        *(uint4*)(sm + el * KPADE + f0) = val;
    }
    __syncthreads();

    const int lane = tid & 31, wave = tid >> 5;
    const int half = lane >> 4, ln = lane & 15;
    v8f acc[4] = {v8f{}, v8f{}, v8f{}, v8f{}};

    for (int kt = 0; kt < KTE; ++kt) {
        v16h b = *(const v16h*)(Wp + (((long)kt * NTILES + wave) * 32 + lane) * 16);
        // A layout: elements 0..7 -> k = base..base+7, 8..15 -> base+16..base+23
        int base = kt * 32 + half * 8;
#pragma unroll
        for (int rt = 0; rt < 4; ++rt) {
            const _Float16* ap = &sm[(rt * 16 + ln) * KPADE + base];
            v8h al = *(const v8h*)ap;
            v8h ah = *(const v8h*)(ap + 16);
            v16h a;
#pragma unroll
            for (int i = 0; i < 8; ++i) { a[i] = al[i]; a[i + 8] = ah[i]; }
            acc[rt] = __builtin_amdgcn_wmma_f32_16x16x32_f16(false, a, false, b,
                                                             (short)0, acc[rt], false, false);
        }
    }

    const int hcol = wave * 16 + ln;
    const float bsv = bias[hcol];
#pragma unroll
    for (int rt = 0; rt < 4; ++rt) {
#pragma unroll
        for (int j = 0; j < 8; ++j) {
            int r = rt * 16 + j + half * 8;            // C/D layout: row, col=lane&15
            int e = e0 + r;
            if (e < E) {
                float s = silu_f(acc[rt][j] + bsv) * s_env[r];
                atomicAdd(&agg[(long)s_col[r] * HDIM + hcol], s);
            }
        }
    }
}

// ---------------------------------------------------------------------------
// Node layer: v += silu(agg @ Wv + bv); agg re-zeroed in place.
// ---------------------------------------------------------------------------
__global__ void __launch_bounds__(256)
node_layer_kernel(float* __restrict__ agg, const _Float16* __restrict__ Wp,
                  const float* __restrict__ bias,
                  float* __restrict__ v, _Float16* __restrict__ vh, int N) {
    __shared__ _Float16 sm[32 * HDIM];                // 8 KB
    const int tid = threadIdx.x;
    const int n0  = blockIdx.x * 32;

    // float4 read + zero, convert to f16
    for (int idx = tid; idx < 32 * (HDIM / 4); idx += 256) {
        int nl = idx >> 5;
        int f0 = (idx & 31) * 4;
        int n  = n0 + nl;
        float4 val = make_float4(0.f, 0.f, 0.f, 0.f);
        if (n < N) {
            float4* gp = (float4*)(agg + (long)n * HDIM + f0);
            val = *gp;
            *gp = make_float4(0.f, 0.f, 0.f, 0.f);    // clean for next layer / replay
        }
        v4h hv; hv[0] = (_Float16)val.x; hv[1] = (_Float16)val.y;
                hv[2] = (_Float16)val.z; hv[3] = (_Float16)val.w;
        *(v4h*)(sm + nl * HDIM + f0) = hv;
    }
    __syncthreads();

    const int lane = tid & 31, wave = tid >> 5;
    const int half = lane >> 4, ln = lane & 15;
    v8f acc0 = {}, acc1 = {};

    for (int kt = 0; kt < KTN; ++kt) {
        v16h b = *(const v16h*)(Wp + (((long)kt * NTILES + wave) * 32 + lane) * 16);
        int base = kt * 32 + half * 8;
        v8h a0l = *(const v8h*)(&sm[ln * HDIM + base]);
        v8h a0h = *(const v8h*)(&sm[ln * HDIM + base + 16]);
        v8h a1l = *(const v8h*)(&sm[(16 + ln) * HDIM + base]);
        v8h a1h = *(const v8h*)(&sm[(16 + ln) * HDIM + base + 16]);
        v16h a0, a1;
#pragma unroll
        for (int i = 0; i < 8; ++i) { a0[i] = a0l[i]; a0[i + 8] = a0h[i];
                                      a1[i] = a1l[i]; a1[i + 8] = a1h[i]; }
        acc0 = __builtin_amdgcn_wmma_f32_16x16x32_f16(false, a0, false, b, (short)0, acc0, false, false);
        acc1 = __builtin_amdgcn_wmma_f32_16x16x32_f16(false, a1, false, b, (short)0, acc1, false, false);
    }

    const int hcol = wave * 16 + ln;
    const float bsv = bias[hcol];
#pragma unroll
    for (int j = 0; j < 8; ++j) {
        int r0 = j + half * 8;
        int n = n0 + r0;
        if (n < N) {
            long gi = (long)n * HDIM + hcol;
            float nv = v[gi] + silu_f(acc0[j] + bsv);
            v[gi] = nv; vh[gi] = (_Float16)nv;
        }
        n = n0 + 16 + r0;
        if (n < N) {
            long gi = (long)n * HDIM + hcol;
            float nv = v[gi] + silu_f(acc1[j] + bsv);
            v[gi] = nv; vh[gi] = (_Float16)nv;
        }
    }
}

// ---------------------------------------------------------------------------
// Head: out = v @ Wh + bh (A sourced from the f16 copy of v)
// ---------------------------------------------------------------------------
__global__ void __launch_bounds__(256)
head_kernel(const _Float16* __restrict__ vh, const _Float16* __restrict__ Wp,
            const float* __restrict__ bias, float* __restrict__ out, int N) {
    __shared__ _Float16 sm[32 * HDIM];
    const int tid = threadIdx.x;
    const int n0  = blockIdx.x * 32;

    for (int idx = tid; idx < 32 * (HDIM / 8); idx += 256) {
        int nl = idx >> 4;
        int f0 = (idx & 15) * 8;
        int n  = n0 + nl;
        uint4 val = make_uint4(0u, 0u, 0u, 0u);
        if (n < N) val = *(const uint4*)(vh + (long)n * HDIM + f0);
        *(uint4*)(sm + nl * HDIM + f0) = val;
    }
    __syncthreads();

    const int lane = tid & 31, wave = tid >> 5;
    const int half = lane >> 4, ln = lane & 15;
    v8f acc0 = {}, acc1 = {};

    for (int kt = 0; kt < KTN; ++kt) {
        v16h b = *(const v16h*)(Wp + (((long)kt * NTILES + wave) * 32 + lane) * 16);
        int base = kt * 32 + half * 8;
        v8h a0l = *(const v8h*)(&sm[ln * HDIM + base]);
        v8h a0h = *(const v8h*)(&sm[ln * HDIM + base + 16]);
        v8h a1l = *(const v8h*)(&sm[(16 + ln) * HDIM + base]);
        v8h a1h = *(const v8h*)(&sm[(16 + ln) * HDIM + base + 16]);
        v16h a0, a1;
#pragma unroll
        for (int i = 0; i < 8; ++i) { a0[i] = a0l[i]; a0[i + 8] = a0h[i];
                                      a1[i] = a1l[i]; a1[i + 8] = a1h[i]; }
        acc0 = __builtin_amdgcn_wmma_f32_16x16x32_f16(false, a0, false, b, (short)0, acc0, false, false);
        acc1 = __builtin_amdgcn_wmma_f32_16x16x32_f16(false, a1, false, b, (short)0, acc1, false, false);
    }

    const int hcol = wave * 16 + ln;
    const float bsv = bias[hcol];
#pragma unroll
    for (int j = 0; j < 8; ++j) {
        int r0 = j + half * 8;
        int n = n0 + r0;
        if (n < N) out[(long)n * HDIM + hcol] = acc0[j] + bsv;
        n = n0 + 16 + r0;
        if (n < N) out[(long)n * HDIM + hcol] = acc1[j] + bsv;
    }
}

// ---------------------------------------------------------------------------
// Host launcher
// ---------------------------------------------------------------------------
extern "C" void kernel_launch(void* const* d_in, const int* in_sizes, int n_in,
                              void* d_out, int out_size, void* d_ws, size_t ws_size,
                              hipStream_t stream) {
    const int*   z       = (const int*)  d_in[0];
    const float* pos     = (const float*)d_in[1];
    const float* cell    = (const float*)d_in[2];
    const int*   eidx    = (const int*)  d_in[3];
    const int*   coff    = (const int*)  d_in[4];
    const int*   ebatch  = (const int*)  d_in[5];
    const float* aemb    = (const float*)d_in[6];
    const float* dtab    = (const float*)d_in[7];
    const float* We      = (const float*)d_in[8];
    const float* be      = (const float*)d_in[9];
    const float* Wv      = (const float*)d_in[10];
    const float* bv      = (const float*)d_in[11];
    const float* Wh      = (const float*)d_in[12];
    const float* bh      = (const float*)d_in[13];

    const int N = in_sizes[0];
    const int E = in_sizes[5];
    const int L = in_sizes[9] / HDIM;                 // 3
    const int KE_SRC = in_sizes[8] / (L * HDIM);      // 306

    char* wp = (char*)d_ws;
    auto carve = [&](size_t bytes) { char* r = wp; wp += (bytes + 255) & ~(size_t)255; return (void*)r; };
    float*    v     = (float*)    carve((size_t)N * HDIM * 4);
    _Float16* vh    = (_Float16*) carve((size_t)N * HDIM * 2);
    float*    agg   = (float*)    carve((size_t)N * HDIM * 4);
    float*    env   = (float*)    carve((size_t)E * 4);
    int*      didx  = (int*)      carve((size_t)E * 4);
    _Float16* dembh = (_Float16*) carve((size_t)RFEAT * RFEAT * 2);
    _Float16* Wep   = (_Float16*) carve((size_t)L * KPADE * HDIM * 2);
    _Float16* Wvp   = (_Float16*) carve((size_t)L * HDIM * HDIM * 2);
    _Float16* Whp   = (_Float16*) carve((size_t)HDIM * HDIM * 2);

    float* out_v    = (float*)d_out;                  // [N,128]
    float* out_pos  = out_v + (size_t)N * HDIM;       // [N,3]
    float* out_dist = out_pos + (size_t)N * 3;        // [E]

    const int* erow = eidx;
    const int* ecol = eidx + E;

    zero_f32_kernel<<<512, 256, 0, stream>>>(agg, N * HDIM);
    for (int i = 0; i < L; ++i) {
        pack_w_kernel<<<(KPADE * HDIM + 255) / 256, 256, 0, stream>>>(
            We + (size_t)i * KE_SRC * HDIM, Wep + (size_t)i * KPADE * HDIM, KE_SRC, KPADE);
        pack_w_kernel<<<(HDIM * HDIM + 255) / 256, 256, 0, stream>>>(
            Wv + (size_t)i * HDIM * HDIM, Wvp + (size_t)i * HDIM * HDIM, HDIM, HDIM);
    }
    pack_w_kernel<<<(HDIM * HDIM + 255) / 256, 256, 0, stream>>>(Wh, Whp, HDIM, HDIM);
    demb_f16_kernel<<<(RFEAT * RFEAT + 255) / 256, 256, 0, stream>>>(dtab, dembh, RFEAT * RFEAT);
    embed_kernel<<<((size_t)N * HDIM + 255) / 256, 256, 0, stream>>>(z, aemb, v, vh, N);
    pos_copy_kernel<<<(N * 3 + 255) / 256, 256, 0, stream>>>(pos, out_pos, N * 3);
    geom_kernel<<<(E + 255) / 256, 256, 0, stream>>>(pos, cell, eidx, coff, ebatch,
                                                     out_dist, env, didx, E);

    const int eg = (E + TEDGE - 1) / TEDGE;
    const int ng = (N + 31) / 32;
    for (int i = 0; i < L; ++i) {
        edge_layer_kernel<<<eg, 256, 0, stream>>>(vh, dembh, didx, env, erow, ecol,
                                                  Wep + (size_t)i * KPADE * HDIM,
                                                  be + (size_t)i * HDIM, agg, E);
        node_layer_kernel<<<ng, 256, 0, stream>>>(agg, Wvp + (size_t)i * HDIM * HDIM,
                                                  bv + (size_t)i * HDIM, v, vh, N);
    }

    head_kernel<<<ng, 256, 0, stream>>>(vh, Whp, bh, out_v, N);
}
